// PatchWiseSynthesisLayer_51384988729651
// MI455X (gfx1250) — matrix operand order repaired and checked by reading, not compiled, not run on hardware
//
#include <hip/hip_runtime.h>
#include <hip/hip_bf16.h>
#include <math.h>

// ---------------------------------------------------------------------------
// PatchWiseSynthesisLayer for MI455X (gfx1250, wave32, WMMA bf16)
//   B=2, CIN=512, COUT=512, RES=256, WDIM=512, K=3, GRID=8, patch width=32
// Conv as implicit GEMM: M=cout, N=spatial, K=cin*9, bf16 WMMA, fp32 acc.
// ---------------------------------------------------------------------------

typedef __bf16 v16bf __attribute__((ext_vector_type(16)));
typedef float  v8f   __attribute__((ext_vector_type(8)));
typedef unsigned int v4u __attribute__((ext_vector_type(4)));

#define B_     2
#define CIN_   512
#define COUT_  512
#define RES_   256
#define WDIM_  512
#define GRID_  8
#define PATCHW 32
#define LRELU  0.2f
#define GAIN   1.4142135623730951f

// force a b128 load on the GLOBAL path (LOADcnt only, never FLAT)
typedef const v4u __attribute__((address_space(1)))* gptr_v4u;
__device__ __forceinline__ v4u gload4(const unsigned int* p) {
  return *(gptr_v4u)(unsigned long long)(uintptr_t)p;
}

// round-to-nearest-even fp32 -> bf16 bit pattern
__device__ __forceinline__ unsigned short f2bf(float f) {
  unsigned int u = __float_as_uint(f);
  unsigned int r = u + 0x7fffu + ((u >> 16) & 1u);
  return (unsigned short)(r >> 16);
}

// ---------------------------------------------------------------------------
// styles[b][s][c] = dot(wg[b][s][:], affine_w[c][:]) / sqrt(WDIM) + affine_b[c]
// ---------------------------------------------------------------------------
__global__ void k_styles(const float* __restrict__ w,
                         const float* __restrict__ wctx,
                         const float* __restrict__ aw,
                         const float* __restrict__ ab,
                         float* __restrict__ styles) {
  const int bs = blockIdx.x;          // b*3 + s
  const int b = bs / 3, s = bs % 3;
  const int c = threadIdx.x;          // 0..511
  const float* wg;
  if (s == 0)      wg = wctx + (size_t)b * 2 * WDIM_;
  else if (s == 1) wg = w + (size_t)b * WDIM_;
  else             wg = wctx + ((size_t)b * 2 + 1) * WDIM_;
  const float* awr = aw + (size_t)c * WDIM_;
  float acc = 0.f;
  for (int d = 0; d < WDIM_; ++d) acc += wg[d] * awr[d];
  styles[(size_t)bs * CIN_ + c] = acc * 0.044194173824159216f + ab[c];
}

// ---------------------------------------------------------------------------
// per-(b, c, patch) unbiased std over 256x32 -> 1/(std + 1e-8)
// ---------------------------------------------------------------------------
__global__ void k_invstd(const float* __restrict__ x, float* __restrict__ inv) {
  const int gid   = blockIdx.x;
  const int patch = gid & 7;
  const int c     = (gid >> 3) & (CIN_ - 1);
  const int b     = gid >> 12;
  const int tid   = threadIdx.x;
  const float* xp = x + (((size_t)(b * CIN_ + c)) * RES_) * RES_ + patch * PATCHW;
  float s1 = 0.f, s2 = 0.f;
  for (int i = tid; i < RES_ * PATCHW; i += 256) {
    float v = xp[(size_t)(i >> 5) * RES_ + (i & 31)];
    s1 += v; s2 += v * v;
  }
  __shared__ float r1[256], r2[256];
  r1[tid] = s1; r2[tid] = s2;
  __syncthreads();
  for (int off = 128; off > 0; off >>= 1) {
    if (tid < off) { r1[tid] += r1[tid + off]; r2[tid] += r2[tid + off]; }
    __syncthreads();
  }
  if (tid == 0) {
    const float n = (float)(RES_ * PATCHW);
    float var = (r2[0] - r1[0] * r1[0] / n) / (n - 1.0f);
    float sd = sqrtf(fmaxf(var, 0.f));
    inv[gid] = 1.0f / (sd + 1e-8f);
  }
}

// ---------------------------------------------------------------------------
// fused multiplier m[b][c][w] = invstd[b][c][w/32] * style_lerp(b, c, w)
// ---------------------------------------------------------------------------
__global__ void k_mmul(const float* __restrict__ styles,
                       const float* __restrict__ inv,
                       const int* __restrict__ lbi,
                       float* __restrict__ m) {
  const int idx = blockIdx.x * blockDim.x + threadIdx.x;
  if (idx >= B_ * CIN_ * RES_) return;
  const int wcol = idx & (RES_ - 1);
  const int c    = (idx >> 8) & (CIN_ - 1);
  const int b    = idx >> 17;
  const float center = (float)(lbi[b] * (RES_ / GRID_));
  float t = ((float)wcol + 0.5f - center) / (float)RES_;   // D = RES*2/2
  t = fminf(fmaxf(t, -1.0f), 1.0f);
  float a = fminf(fmaxf(-t, 0.0f), 1.0f);
  float r = fminf(fmaxf(t, 0.0f), 1.0f);
  float mid = 1.0f - a - r;
  const float s0 = styles[((size_t)b * 3 + 0) * CIN_ + c];
  const float s1 = styles[((size_t)b * 3 + 1) * CIN_ + c];
  const float s2 = styles[((size_t)b * 3 + 2) * CIN_ + c];
  float scol = a * s0 + mid * s1 + r * s2;
  m[idx] = scol * inv[((size_t)b * CIN_ + c) * GRID_ + (wcol >> 5)];
}

// ---------------------------------------------------------------------------
// weights fp32 [cout][cin][3][3] -> bf16 pairs, uint wb[kk][cout][cin/2]
// ---------------------------------------------------------------------------
__global__ void k_wb(const float* __restrict__ weight, unsigned int* __restrict__ wb) {
  const int idx = blockIdx.x * blockDim.x + threadIdx.x;
  if (idx >= 9 * COUT_ * (CIN_ / 2)) return;
  const int cp   = idx & 255;
  const int cout = (idx >> 8) & (COUT_ - 1);
  const int kk   = idx >> 17;
  const float lo = weight[((size_t)cout * CIN_ + 2 * cp) * 9 + kk];
  const float hi = weight[((size_t)cout * CIN_ + 2 * cp + 1) * 9 + kk];
  wb[idx] = (unsigned int)f2bf(lo) | ((unsigned int)f2bf(hi) << 16);
}

// ---------------------------------------------------------------------------
// demod[cout] = rsqrt(sum w^2 + 1e-8)
// ---------------------------------------------------------------------------
__global__ void k_demod(const float* __restrict__ weight, float* __restrict__ demod) {
  const int cout = blockIdx.x, tid = threadIdx.x;
  const float* wp = weight + (size_t)cout * CIN_ * 9;
  float s = 0.f;
  for (int i = tid; i < CIN_ * 9; i += 256) { float v = wp[i]; s += v * v; }
  __shared__ float r[256];
  r[tid] = s; __syncthreads();
  for (int off = 128; off > 0; off >>= 1) {
    if (tid < off) r[tid] += r[tid + off];
    __syncthreads();
  }
  if (tid == 0) demod[cout] = rsqrtf(r[0] + 1e-8f);
}

// ---------------------------------------------------------------------------
// channels-last modulated bf16 activations:
//   xmu[((b*RES+h)*RES+w)*256 + cp] = pack_bf16(x*m at cin=2cp, 2cp+1)
// ---------------------------------------------------------------------------
__global__ __launch_bounds__(256) void k_xmod(const float* __restrict__ x,
                                              const float* __restrict__ m,
                                              unsigned int* __restrict__ xmu) {
  __shared__ unsigned short tile[64][33];
  const int bh = blockIdx.x;                 // b*RES + h
  const int b  = bh >> 8, h = bh & 255;
  const int c0 = blockIdx.y * 64;
  const int w0 = blockIdx.z * 32;
  const int tid = threadIdx.x;
  const int wl = tid & 31;
  const int clBase = tid >> 5;               // 0..7
#pragma unroll
  for (int j = 0; j < 8; ++j) {
    const int cl = clBase + j * 8;
    const int c = c0 + cl;
    const float v = x[((size_t)(b * CIN_ + c) * RES_ + h) * RES_ + w0 + wl]
                  * m[(size_t)(b * CIN_ + c) * RES_ + w0 + wl];
    tile[cl][wl] = f2bf(v);
  }
  __syncthreads();
  const int cp = tid & 31;
  const int wlb = tid >> 5;
#pragma unroll
  for (int j = 0; j < 4; ++j) {
    const int wloc = wlb + j * 8;
    const unsigned int d = (unsigned int)tile[2 * cp][wloc]
                         | ((unsigned int)tile[2 * cp + 1][wloc] << 16);
    xmu[((size_t)(b * RES_ + h) * RES_ + w0 + wloc) * (CIN_ / 2) + (c0 >> 1) + cp] = d;
  }
}

// ---------------------------------------------------------------------------
// FAST conv: block tile 128 cout x 128 spatial (4 rows x 32 cols),
// wave tile 32 x 64 -> 8 WMMA per K-step. AS1 loads (never FLAT),
// waves_per_eu(1) to give the allocator the full VGPR budget (no spills),
// double-buffered LDS, one barrier per K-step, incremental u32 offsets.
// grid = 16 patches * 4 cout-tiles * 64 spatial-tiles = 4096 blocks of 256.
// ---------------------------------------------------------------------------
__global__ __launch_bounds__(256)
__attribute__((amdgpu_waves_per_eu(1)))
void k_conv_fast(
    const unsigned int* __restrict__ xmu, const unsigned int* __restrict__ wb,
    const float* __restrict__ demod, const float* __restrict__ bias,
    const float* __restrict__ noise, const float* __restrict__ nsptr,
    float* __restrict__ out) {
  __shared__ unsigned int ldsW[2][128 * 20];  // 128 couts x 16 K-pair dwords (+4 pad)
  __shared__ unsigned int ldsX[2][128 * 20];  // 128 spatial x 16 K-pair dwords (+4 pad)

  const int tid = threadIdx.x;
  int bi = blockIdx.x;
  const int sT    = bi & 63;  bi >>= 6;
  const int coT   = bi & 3;   bi >>= 2;
  const int patch = bi & 7;
  const int b     = bi >> 3;
  const int h0    = sT * 4;
  const int cout0 = coT * 128;

  const int lane = tid & 31;
  const int wid  = tid >> 5;
  const int wm   = wid >> 1;            // 0..3 -> 32-cout sub-tile
  const int wn   = wid & 1;             // 0..1 -> 64-spatial sub-tile
  const int ml   = lane & 15;
  const int half = lane >> 4;

  // W loader: thread -> (row 0..127, 8-dword half)
  const int wRow = tid >> 1;
  const int wDwb = (tid & 1) * 8;
  // X loader: thread -> (n 0..127, 8-dword half)
  const int xN   = tid >> 1;
  const int xDwb = (tid & 1) * 8;
  const int xRow = xN >> 5;             // 0..3
  const int xCol = xN & 31;

  v8f acc[2][4] = {};

  // u32 element offsets (xmu: 33.5M dwords, wb: 1.18M dwords -> fit u32)
  unsigned int woff = (unsigned int)(cout0 + wRow) * (CIN_ / 2) + wDwb;  // kk=0
  unsigned int xoff;
  bool valid;
  int kh = 0, kw = 0;
  auto xrebase = [&]() {
    const int hs  = h0 + xRow + kh - 1;
    const int wsc = xCol + kw - 1;
    valid = (hs >= 0) & (hs < RES_) & (wsc >= 0) & (wsc < PATCHW);
    const int hsc  = valid ? hs : 0;
    const int wscc = valid ? wsc : 0;
    xoff = (unsigned int)((b * RES_ + hsc) * RES_ + patch * PATCHW + wscc)
               * (CIN_ / 2) + xDwb;
  };
  xrebase();

  v4u fw0, fw1, fx0, fx1;
  auto fetch = [&]() {
    fw0 = gload4(wb + woff);
    fw1 = gload4(wb + woff + 4);
    if (valid) {
      fx0 = gload4(xmu + xoff);
      fx1 = gload4(xmu + xoff + 4);
    } else {
      const v4u z = {0u, 0u, 0u, 0u};
      fx0 = z;
      fx1 = z;
    }
  };
  fetch();

  for (int it = 0; it < 144; ++it) {
    const int buf = it & 1;
    {
      v4u* w4 = (v4u*)&ldsW[buf][wRow * 20 + wDwb];
      w4[0] = fw0;
      w4[1] = fw1;
      v4u* x4 = (v4u*)&ldsX[buf][xN * 20 + xDwb];
      x4[0] = fx0;
      x4[1] = fx1;
    }
    if (it != 143) {                    // prepare next stage (overlaps compute)
      if ((it & 15) == 15) {            // cin chunks exhausted -> next 3x3 tap
        ++kw;
        if (kw == 3) { kw = 0; ++kh; }
        const int kk = kh * 3 + kw;
        woff = (unsigned int)(kk * COUT_ + cout0 + wRow) * (CIN_ / 2) + wDwb;
        xrebase();
      } else {
        woff += 16;
        xoff += 16;
      }
      fetch();
    }
    __syncthreads();

    union { v16bf v; unsigned int u[8]; } A0, A1, Bv[4];
    const int r0 = wm * 32 + ml;
#pragma unroll
    for (int d = 0; d < 8; ++d) {
      const int p = half * 8 + d;  // contiguous K-pair map (k = 16*half + s)
      A0.u[d] = ldsW[buf][r0 * 20 + p];
      A1.u[d] = ldsW[buf][(r0 + 16) * 20 + p];
    }
#pragma unroll
    for (int s = 0; s < 4; ++s) {
      const int n = wn * 64 + s * 16 + ml;
#pragma unroll
      for (int d = 0; d < 8; ++d)
        Bv[s].u[d] = ldsX[buf][n * 20 + half * 8 + d];    // contiguous K-pair map, same as A
    }
#pragma unroll
    for (int s = 0; s < 4; ++s) {
      acc[0][s] = __builtin_amdgcn_wmma_f32_16x16x32_bf16(
          false, A0.v, false, Bv[s].v, (short)0, acc[0][s], false, false);
      acc[1][s] = __builtin_amdgcn_wmma_f32_16x16x32_bf16(
          false, A1.v, false, Bv[s].v, (short)0, acc[1][s], false, false);
    }
  }

  // ---- epilogue: demod, noise fma, bias, lrelu * sqrt(2) ----
  const float ns = nsptr[0];
#pragma unroll
  for (int sm = 0; sm < 2; ++sm)
#pragma unroll
    for (int sn = 0; sn < 4; ++sn)
#pragma unroll
      for (int v = 0; v < 8; ++v) {
        const int co = cout0 + wm * 32 + sm * 16 + v + half * 8;  // C-frag M
        const int n  = wn * 64 + sn * 16 + ml;                    // C-frag N
        const int s  = sT * 128 + n;
        const int h  = s >> 5;
        const int wg = patch * PATCHW + (s & 31);
        float r = acc[sm][sn][v] * demod[co]
                + noise[((size_t)b * RES_ + h) * RES_ + wg] * ns + bias[co];
        r = (r >= 0.f ? r : r * LRELU) * GAIN;
        out[(((size_t)b * COUT_ + co) * RES_ + h) * RES_ + wg] = r;
      }
}

// ---------------------------------------------------------------------------
// FALLBACK conv (small workspace): on-the-fly modulation loader.
// ---------------------------------------------------------------------------
__global__ __launch_bounds__(256) void k_conv_fallback(
    const float* __restrict__ x, const unsigned int* __restrict__ wb,
    const float* __restrict__ mmul, const float* __restrict__ demod,
    const float* __restrict__ bias, const float* __restrict__ noise,
    const float* __restrict__ nsptr, float* __restrict__ out) {
  __shared__ unsigned int ldsW[128 * 20];
  __shared__ unsigned int ldsX[64 * 20];

  const int tid = threadIdx.x;
  int bi = blockIdx.x;
  const int sT    = bi & 127; bi >>= 7;
  const int coT   = bi & 3;   bi >>= 2;
  const int patch = bi & 7;
  const int b     = bi >> 3;
  const int h0    = sT * 2;
  const int cout0 = coT * 128;

  const int lane = tid & 31;
  const int wid  = tid >> 5;
  const int wm   = wid >> 1;
  const int wn   = wid & 1;
  const int ml   = lane & 15;
  const int half = lane >> 4;

  const int wRow = tid >> 1;
  const int wDwb = (tid & 1) * 8;
  const int xK   = tid >> 3;
  const int xNb  = (tid & 7) * 8;
  const int xRow = xNb >> 5;
  const int xWc  = xNb & 31;

  v8f acc[2][2] = {};

  for (int kk = 0; kk < 9; ++kk) {
    const int kh = kk / 3, kw = kk % 3;
    const int hs = h0 + xRow + kh - 1;
    const bool rowOK = (hs >= 0) && (hs < RES_);
    for (int c0 = 0; c0 < CIN_; c0 += 32) {
      const unsigned int* wsrc =
          wb + ((size_t)kk * COUT_ + cout0 + wRow) * (CIN_ / 2) + (c0 >> 1) + wDwb;
#pragma unroll
      for (int i = 0; i < 8; ++i) ldsW[wRow * 20 + wDwb + i] = wsrc[i];
      {
        const int cin = c0 + xK;
        const float* xrow = x + ((size_t)(b * CIN_ + cin) * RES_ + (rowOK ? hs : 0)) * RES_
                              + patch * PATCHW;
        const float* mrow = mmul + (size_t)(b * CIN_ + cin) * RES_ + patch * PATCHW;
        unsigned short* xl = (unsigned short*)ldsX;
#pragma unroll
        for (int i = 0; i < 8; ++i) {
          const int wc  = xWc + i;
          const int wsc = wc + kw - 1;
          float v = 0.f;
          if (rowOK && wsc >= 0 && wsc < PATCHW) v = xrow[wsc] * mrow[wsc];
          xl[(xNb + i) * 40 + xK] = f2bf(v);
        }
      }
      __syncthreads();

      union { v16bf v; unsigned int u[8]; } A0, A1, Bv0, Bv1;
      const int r0 = wm * 32 + ml;
#pragma unroll
      for (int d = 0; d < 8; ++d) {
        const int p = half * 8 + d;
        A0.u[d] = ldsW[r0 * 20 + p];
        A1.u[d] = ldsW[(r0 + 16) * 20 + p];
      }
      const int n0 = wn * 32 + ml;
#pragma unroll
      for (int d = 0; d < 8; ++d) {
        Bv0.u[d] = ldsX[n0 * 20 + half * 8 + d];
        Bv1.u[d] = ldsX[(n0 + 16) * 20 + half * 8 + d];
      }
      acc[0][0] = __builtin_amdgcn_wmma_f32_16x16x32_bf16(
          false, A0.v, false, Bv0.v, (short)0, acc[0][0], false, false);
      acc[0][1] = __builtin_amdgcn_wmma_f32_16x16x32_bf16(
          false, A0.v, false, Bv1.v, (short)0, acc[0][1], false, false);
      acc[1][0] = __builtin_amdgcn_wmma_f32_16x16x32_bf16(
          false, A1.v, false, Bv0.v, (short)0, acc[1][0], false, false);
      acc[1][1] = __builtin_amdgcn_wmma_f32_16x16x32_bf16(
          false, A1.v, false, Bv1.v, (short)0, acc[1][1], false, false);
      __syncthreads();
    }
  }

  const float ns = nsptr[0];
#pragma unroll
  for (int sm = 0; sm < 2; ++sm)
#pragma unroll
    for (int sn = 0; sn < 2; ++sn)
#pragma unroll
      for (int v = 0; v < 8; ++v) {
        const int co = cout0 + wm * 32 + sm * 16 + v + half * 8;
        const int n  = wn * 32 + sn * 16 + ml;
        const int s  = sT * 64 + n;
        const int h  = s >> 5;
        const int wg = patch * PATCHW + (s & 31);
        float r = acc[sm][sn][v] * demod[co]
                + noise[((size_t)b * RES_ + h) * RES_ + wg] * ns + bias[co];
        r = (r >= 0.f ? r : r * LRELU) * GAIN;
        out[(((size_t)b * COUT_ + co) * RES_ + h) * RES_ + wg] = r;
      }
}

// ---------------------------------------------------------------------------
// launch
// ---------------------------------------------------------------------------
extern "C" void kernel_launch(void* const* d_in, const int* in_sizes, int n_in,
                              void* d_out, int out_size, void* d_ws, size_t ws_size,
                              hipStream_t stream) {
  const float* x      = (const float*)d_in[0];
  const float* w      = (const float*)d_in[1];
  const float* wctx   = (const float*)d_in[2];
  const float* weight = (const float*)d_in[3];
  const float* aw     = (const float*)d_in[4];
  const float* ab     = (const float*)d_in[5];
  const float* bias   = (const float*)d_in[6];
  const float* nstr   = (const float*)d_in[7];
  const float* noise  = (const float*)d_in[8];
  const int*   lbi    = (const int*)d_in[9];
  float* out = (float*)d_out;

  // workspace layout (dwords)
  const size_t N_STYLES = 3072;            // 2*3*512
  const size_t N_INV    = 8192;            // 2*512*8
  const size_t N_M      = 262144;          // 2*512*256
  const size_t N_DEMOD  = 512;
  const size_t N_WB     = 1179648;         // 9*512*256
  const size_t N_XMU    = (size_t)B_ * RES_ * RES_ * (CIN_ / 2);  // 33554432

  float* ws            = (float*)d_ws;
  float* ws_styles     = ws;
  float* ws_inv        = ws_styles + N_STYLES;
  float* ws_m          = ws_inv + N_INV;
  float* ws_demod      = ws_m + N_M;
  unsigned int* ws_wb  = (unsigned int*)(ws_demod + N_DEMOD);
  unsigned int* ws_xmu = ws_wb + N_WB;

  const size_t need_small = (N_STYLES + N_INV + N_M + N_DEMOD + N_WB) * 4;
  const size_t need_big   = need_small + N_XMU * 4;

  k_styles<<<B_ * 3, 512, 0, stream>>>(w, wctx, aw, ab, ws_styles);
  k_invstd<<<B_ * CIN_ * GRID_, 256, 0, stream>>>(x, ws_inv);
  k_mmul<<<(B_ * CIN_ * RES_ + 255) / 256, 256, 0, stream>>>(ws_styles, ws_inv, lbi, ws_m);
  k_wb<<<(9 * COUT_ * (CIN_ / 2) + 255) / 256, 256, 0, stream>>>(weight, ws_wb);
  k_demod<<<COUT_, 256, 0, stream>>>(weight, ws_demod);

  if (ws_size >= need_big) {
    dim3 g(B_ * RES_, CIN_ / 64, RES_ / 32);
    k_xmod<<<g, 256, 0, stream>>>(x, ws_m, ws_xmu);
    k_conv_fast<<<16 * 4 * 64, 256, 0, stream>>>(ws_xmu, ws_wb, ws_demod, bias,
                                                 noise, nstr, out);
  } else {
    k_conv_fallback<<<16 * 4 * 128, 256, 0, stream>>>(x, ws_wb, ws_m, ws_demod, bias,
                                                      noise, nstr, out);
  }
}
